// MultiheadAttention_90572270338524
// MI455X (gfx1250) — compile-verified, matrix operand order checked by reading, not run on hardware
//
#include <hip/hip_runtime.h>
#include <hip/hip_bf16.h>

#define T_DIM 2048
#define B_DIM 2
#define E_DIM 256
#define H_DIM 8
#define D_DIM 32
#define M_ROWS (T_DIM * B_DIM)   // 4096 rows of the (t,b) flattened GEMMs

typedef __attribute__((ext_vector_type(16))) __bf16 v16bf;
typedef __attribute__((ext_vector_type(8)))  __bf16 v8bf;
typedef __attribute__((ext_vector_type(8)))  float  v8f;

// ---------------------------------------------------------------------------
// Fragment loader for 16-bit A/B WMMA operands. Per CDNA5 ISA:
//   lane l (l<16):  row M = l,    K = kbase..kbase+7 (V0-3), kbase+16..+23 (V4-7)
//   lane l (l>=16): row M = l-16, kbase = 8
// Caller passes row = lane&15 and kcol = ...+8*(lane>>4). Two 16B loads/lane.
// ---------------------------------------------------------------------------
__device__ __forceinline__ v16bf load_frag(const __bf16* base, int row, int kcol,
                                           int stride) {
  const __bf16* p = base + (size_t)row * (size_t)stride + kcol;
  v8bf lo = *(const v8bf*)(p);
  v8bf hi = *(const v8bf*)(p + 16);
  v16bf r;
#pragma unroll
  for (int i = 0; i < 8; ++i) { r[i] = lo[i]; r[i + 8] = hi[i]; }
  return r;
}

__device__ __forceinline__ v8f wmma_bf16(v16bf a, v16bf b, v8f c) {
  return __builtin_amdgcn_wmma_f32_16x16x32_bf16(
      /*neg_a=*/false, a, /*neg_b=*/false, b,
      /*c_mod=*/(short)0, c, /*reuse_a=*/false, /*reuse_b=*/false);
}

// ---------------------------------------------------------------------------
// f32 -> bf16 convert
// ---------------------------------------------------------------------------
__global__ __launch_bounds__(256) void cvt_bf16(const float* __restrict__ src,
                                                __bf16* __restrict__ dst, int n) {
  int i = blockIdx.x * blockDim.x + threadIdx.x;
  if (i < n) dst[i] = (__bf16)src[i];
}

// ---------------------------------------------------------------------------
// QKV projection: x[4096,256] @ W{q,k,v}.T + b.  blockIdx.y selects the
// projection (uniform, scalar-branch epilogue). One wave = one 16-row tile,
// looping over all 16 N-tiles. Q written scaled by 1/sqrt(D) as [b,h,t,d];
// K as [b,h,s,d]; V transposed as [b,h,d,s] so PV B-fragments are contiguous.
// ---------------------------------------------------------------------------
__global__ __launch_bounds__(256) void qkv_proj(
    const __bf16* __restrict__ xb,
    const __bf16* __restrict__ Wqb, const __bf16* __restrict__ Wkb,
    const __bf16* __restrict__ Wvb,
    const float* __restrict__ bq, const float* __restrict__ bk,
    const float* __restrict__ bv,
    __bf16* __restrict__ Qb, __bf16* __restrict__ Kb, __bf16* __restrict__ Vt) {
  const int wid  = threadIdx.x >> 5;
  const int lane = threadIdx.x & 31;
  const int hf   = lane >> 4;
  const int l16  = lane & 15;
  const int kb   = hf * 8;

  const int rt   = blockIdx.x * 8 + wid;      // 0..255 row tile
  const int proj = blockIdx.y;                // 0=q 1=k 2=v (uniform)

  const __bf16* W    = (proj == 0) ? Wqb : (proj == 1) ? Wkb : Wvb;
  const float*  bias = (proj == 0) ? bq  : (proj == 1) ? bk  : bv;

  // Preload the 8 A-fragments covering K = 0..255 for this 16-row tile.
  const __bf16* xrow = xb + (size_t)rt * 16 * E_DIM;
  v16bf a[8];
#pragma unroll
  for (int k = 0; k < 8; ++k) a[k] = load_frag(xrow, l16, k * 32 + kb, E_DIM);

  const float scaling = 0.17677669529663687f;  // 1/sqrt(32)

  for (int nt = 0; nt < 16; ++nt) {
    v8f c = {};
#pragma unroll
    for (int k = 0; k < 8; ++k) {
      // B[k=j][n=i] = W[i][j]; column i of B = row i of W (contiguous).
      v16bf bf = load_frag(W, nt * 16 + l16, k * 32 + kb, E_DIM);
      c = wmma_bf16(a[k], bf, c);
    }
    const int i  = nt * 16 + l16;   // output feature
    const int h  = i >> 5;
    const int d  = i & 31;
    const float bi = bias[i];
#pragma unroll
    for (int r = 0; r < 8; ++r) {
      const int row = rt * 16 + r + 8 * hf;   // row = t*B + b
      const int t = row >> 1;
      const int b = row & 1;
      const float val = c[r] + bi;
      if (proj == 0) {
        Qb[(((size_t)(b * H_DIM + h) * T_DIM) + t) * D_DIM + d] =
            (__bf16)(val * scaling);
      } else if (proj == 1) {
        Kb[(((size_t)(b * H_DIM + h) * T_DIM) + t) * D_DIM + d] = (__bf16)val;
      } else {
        Vt[(((size_t)(b * H_DIM + h) * D_DIM) + d) * T_DIM + t] = (__bf16)val;
      }
    }
  }
}

// ---------------------------------------------------------------------------
// Flash attention: one wave owns a 32-row (t) tile of one (b,h) — two 16-row
// subtiles j=0,1 sharing the K/V fragments — and streams S in chunks of 32
// with online softmax. Per chunk: 4 score WMMAs + 4 PV WMMAs.
// P goes through LDS to convert C-fragment layout -> A-fragment layout.
// ---------------------------------------------------------------------------
__global__ __launch_bounds__(256) void attn_flash(
    const __bf16* __restrict__ Qb, const __bf16* __restrict__ Kb,
    const __bf16* __restrict__ Vt, const float* __restrict__ bias,
    __bf16* __restrict__ attnOut) {
  __shared__ __bf16 ptile[8][2][16 * 32];   // 16 KB

  const int wid  = threadIdx.x >> 5;
  const int lane = threadIdx.x & 31;
  const int hf   = lane >> 4;
  const int l16  = lane & 15;
  const int kb   = hf * 8;

  const int id    = blockIdx.x * 8 + wid;   // 0..1023
  const int bh    = id >> 6;                // 0..15  (b*H + h)
  const int ttile = id & 63;                // 0..63
  const int b     = bh >> 3;
  const int h     = bh & 7;
  const int t0    = ttile * 32;

  const __bf16* Qhead = Qb + (size_t)bh * T_DIM * D_DIM;
  const __bf16* Khead = Kb + (size_t)bh * T_DIM * D_DIM;
  const __bf16* Vhead = Vt + (size_t)bh * D_DIM * T_DIM;
  const float*  biasB = bias + (size_t)b * T_DIM * T_DIM;

  // Q A-fragments for both 16-row subtiles, loaded once.
  v16bf qa[2];
#pragma unroll
  for (int j = 0; j < 2; ++j)
    qa[j] = load_frag(Qhead + (size_t)(t0 + 16 * j) * D_DIM, l16, kb, D_DIM);

  float m[2][8], lsum[2][8];
#pragma unroll
  for (int j = 0; j < 2; ++j)
#pragma unroll
    for (int r = 0; r < 8; ++r) { m[j][r] = -1e30f; lsum[j][r] = 0.0f; }
  v8f o[2][2] = {};   // [row subtile][d half]

  for (int s0 = 0; s0 < T_DIM; s0 += 32) {
    // Prefetch next chunk's K/V rows (global_prefetch_b8).
    if (s0 + 32 < T_DIM) {
      __builtin_prefetch(Khead + (size_t)(s0 + 32 + l16) * D_DIM, 0, 3);
      __builtin_prefetch(Vhead + (size_t)l16 * T_DIM + s0 + 32, 0, 3);
      __builtin_prefetch(Vhead + (size_t)(16 + l16) * T_DIM + s0 + 32, 0, 3);
    }

    // K^T B-fragments: column n = sequence position s0+l16 (+16).
    v16bf kf0 = load_frag(Khead + (size_t)s0 * D_DIM,        l16, kb, D_DIM);
    v16bf kf1 = load_frag(Khead + (size_t)(s0 + 16) * D_DIM, l16, kb, D_DIM);

    v8f c[2][2];
#pragma unroll
    for (int j = 0; j < 2; ++j) {
      v8f z0 = {}, z1 = {};
      c[j][0] = wmma_bf16(qa[j], kf0, z0);
      c[j][1] = wmma_bf16(qa[j], kf1, z1);
    }

    // additive attention bias (broadcast across heads) + online softmax
#pragma unroll
    for (int j = 0; j < 2; ++j) {
      float p0[8], p1[8];
#pragma unroll
      for (int r = 0; r < 8; ++r) {
        const int t = t0 + 16 * j + r + 8 * hf;
        const float* bp = biasB + (size_t)t * T_DIM + s0 + l16;
        c[j][0][r] += bp[0];
        c[j][1][r] += bp[16];
      }
#pragma unroll
      for (int r = 0; r < 8; ++r) {
        float rm = fmaxf(c[j][0][r], c[j][1][r]);
        rm = fmaxf(rm, __shfl_xor(rm, 1));
        rm = fmaxf(rm, __shfl_xor(rm, 2));
        rm = fmaxf(rm, __shfl_xor(rm, 4));
        rm = fmaxf(rm, __shfl_xor(rm, 8));
        const float mn    = fmaxf(m[j][r], rm);
        const float scale = __expf(m[j][r] - mn);
        const float e0 = __expf(c[j][0][r] - mn);
        const float e1 = __expf(c[j][1][r] - mn);
        float rs = e0 + e1;
        rs += __shfl_xor(rs, 1);
        rs += __shfl_xor(rs, 2);
        rs += __shfl_xor(rs, 4);
        rs += __shfl_xor(rs, 8);
        lsum[j][r] = lsum[j][r] * scale + rs;
        o[j][0][r] *= scale;
        o[j][1][r] *= scale;
        m[j][r] = mn;
        p0[r] = e0;
        p1[r] = e1;
      }
      // C-layout -> A-layout via LDS (bf16 16x32 tile per subtile)
      __bf16* pt = &ptile[wid][j][0];
#pragma unroll
      for (int r = 0; r < 8; ++r) {
        const int row = r + 8 * hf;
        pt[row * 32 + l16]      = (__bf16)p0[r];
        pt[row * 32 + 16 + l16] = (__bf16)p1[r];
      }
    }
    asm volatile("s_wait_dscnt 0" ::: "memory");

    // V^T B-fragments: column n = d (row of Vt, contiguous in s).
    v16bf vb0 = load_frag(Vhead + s0,              l16, kb, T_DIM);
    v16bf vb1 = load_frag(Vhead + 16 * T_DIM + s0, l16, kb, T_DIM);

#pragma unroll
    for (int j = 0; j < 2; ++j) {
      v16bf pa = load_frag(&ptile[wid][j][0], l16, kb, 32);
      o[j][0] = wmma_bf16(pa, vb0, o[j][0]);
      o[j][1] = wmma_bf16(pa, vb1, o[j][1]);
    }
  }

  // normalize and store attn output as bf16 in [t,b,e] layout
#pragma unroll
  for (int j = 0; j < 2; ++j) {
#pragma unroll
    for (int r = 0; r < 8; ++r) {
      const float inv = 1.0f / lsum[j][r];
      const int t = t0 + 16 * j + r + 8 * hf;
      const size_t off = ((size_t)t * B_DIM + b) * E_DIM + h * D_DIM;
      attnOut[off + l16]      = (__bf16)(o[j][0][r] * inv);
      attnOut[off + 16 + l16] = (__bf16)(o[j][1][r] * inv);
    }
  }
}

// ---------------------------------------------------------------------------
// Output projection: attn[4096,256] @ Wo.T + bo -> f32 out [T,B,E]
// ---------------------------------------------------------------------------
__global__ __launch_bounds__(256) void out_proj(
    const __bf16* __restrict__ ab, const __bf16* __restrict__ Wob,
    const float* __restrict__ bo, float* __restrict__ out) {
  const int wid  = threadIdx.x >> 5;
  const int lane = threadIdx.x & 31;
  const int hf   = lane >> 4;
  const int l16  = lane & 15;
  const int kb   = hf * 8;
  const int rt   = blockIdx.x * 8 + wid;  // 0..255

  const __bf16* arow = ab + (size_t)rt * 16 * E_DIM;
  v16bf a[8];
#pragma unroll
  for (int k = 0; k < 8; ++k) a[k] = load_frag(arow, l16, k * 32 + kb, E_DIM);

  for (int nt = 0; nt < 16; ++nt) {
    v8f c = {};
#pragma unroll
    for (int k = 0; k < 8; ++k) {
      v16bf bf = load_frag(Wob, nt * 16 + l16, k * 32 + kb, E_DIM);
      c = wmma_bf16(a[k], bf, c);
    }
    const int i = nt * 16 + l16;
    const float bi = bo[i];
#pragma unroll
    for (int r = 0; r < 8; ++r) {
      const int row = rt * 16 + r + 8 * hf;
      out[(size_t)row * E_DIM + i] = c[r] + bi;
    }
  }
}

// ---------------------------------------------------------------------------
// Host-side launcher
// ---------------------------------------------------------------------------
extern "C" void kernel_launch(void* const* d_in, const int* in_sizes, int n_in,
                              void* d_out, int out_size, void* d_ws,
                              size_t ws_size, hipStream_t stream) {
  (void)in_sizes; (void)n_in; (void)out_size; (void)ws_size;

  const float* query = (const float*)d_in[0];
  const float* bias  = (const float*)d_in[1];
  const float* Wq    = (const float*)d_in[2];
  const float* bq    = (const float*)d_in[3];
  const float* Wk    = (const float*)d_in[4];
  const float* bk    = (const float*)d_in[5];
  const float* Wv    = (const float*)d_in[6];
  const float* bv    = (const float*)d_in[7];
  const float* Wo    = (const float*)d_in[8];
  const float* bo    = (const float*)d_in[9];
  float* out = (float*)d_out;

  // workspace layout (bytes)
  char* ws = (char*)d_ws;
  const size_t XB_BYTES = (size_t)M_ROWS * E_DIM * 2;   // 2 MB
  const size_t W_BYTES  = (size_t)E_DIM * E_DIM * 2;    // 128 KB each
  const size_t H_BYTES  = (size_t)B_DIM * H_DIM * T_DIM * D_DIM * 2;  // 2 MB

  __bf16* xb  = (__bf16*)(ws);
  __bf16* Wqb = (__bf16*)(ws + XB_BYTES);
  __bf16* Wkb = (__bf16*)(ws + XB_BYTES + 1 * W_BYTES);
  __bf16* Wvb = (__bf16*)(ws + XB_BYTES + 2 * W_BYTES);
  __bf16* Wob = (__bf16*)(ws + XB_BYTES + 3 * W_BYTES);
  __bf16* Qb  = (__bf16*)(ws + XB_BYTES + 4 * W_BYTES);
  __bf16* Kb  = (__bf16*)(ws + XB_BYTES + 4 * W_BYTES + 1 * H_BYTES);
  __bf16* Vt  = (__bf16*)(ws + XB_BYTES + 4 * W_BYTES + 2 * H_BYTES);
  __bf16* Ab  = (__bf16*)(ws + XB_BYTES + 4 * W_BYTES + 3 * H_BYTES);

  const int nQuery = M_ROWS * E_DIM;   // 1048576
  const int nW     = E_DIM * E_DIM;    // 65536

  cvt_bf16<<<(nQuery + 255) / 256, 256, 0, stream>>>(query, xb, nQuery);
  cvt_bf16<<<(nW + 255) / 256, 256, 0, stream>>>(Wq, Wqb, nW);
  cvt_bf16<<<(nW + 255) / 256, 256, 0, stream>>>(Wk, Wkb, nW);
  cvt_bf16<<<(nW + 255) / 256, 256, 0, stream>>>(Wv, Wvb, nW);
  cvt_bf16<<<(nW + 255) / 256, 256, 0, stream>>>(Wo, Wob, nW);

  // 256 row-tiles (8 waves/block) x 3 projections in grid.y
  qkv_proj<<<dim3(32, 3), 256, 0, stream>>>(xb, Wqb, Wkb, Wvb, bq, bk, bv,
                                            Qb, Kb, Vt);

  // B*H*(T/32) = 1024 wave-units, 8 waves/block
  attn_flash<<<128, 256, 0, stream>>>(Qb, Kb, Vt, bias, Ab);

  // 256 row-tiles, 8 waves/block
  out_proj<<<32, 256, 0, stream>>>(Ab, Wob, bo, out);
}
